// AttentionWordRNN_38525856645857
// MI455X (gfx1250) — matrix operand order repaired and checked by reading, not compiled, not run on hardware
//
#include <hip/hip_runtime.h>
#include <hip/hip_bf16.h>

typedef __attribute__((ext_vector_type(16))) __bf16 v16bf;
typedef __attribute__((ext_vector_type(8)))  float  v8f;

#define T_LEN 512
#define B_SZ  64
#define E_DIM 256
#define C_DIM 128
#define K_IN  384      // E + CDIM
#define H_DIM 512
#define G4    2048     // 4*H
#define K_C   896      // H + K_IN (combined K for recurrent GEMM)
#define KT_C  28       // K_C / 32
#define TWOH  1024
#define NC    5
#define NBLK  128      // persistent scan workgroups (1024 waves = 1024 tiles)

union FragU { v16bf v; uint4 u[2]; };

__device__ __forceinline__ float sigf(float x) { return 1.0f / (1.0f + expf(-x)); }

// ---------------------------------------------------------------------------
// Pack recurrent weights into per-lane WMMA B-fragment order (bf16):
//   wpack[(((dir*128 + ntile)*28 + kt)*32 + lane)*16 + e]
// logical: n = ntile*16 + lane%16 ; k = kt*32 + (lane/16)*8 + (e%8) + (e/8)*16
// value   = k<512 ? w_hh[n][k] : w_ih[n][k-512]
// Each B fragment then loads as two global_load_b128 per lane.
// ---------------------------------------------------------------------------
__global__ __launch_bounds__(256) void pack_wcomb_kernel(
    const float* __restrict__ w_ih_f, const float* __restrict__ w_hh_f,
    const float* __restrict__ w_ih_b, const float* __restrict__ w_hh_b,
    __bf16* __restrict__ wpack)
{
    size_t idx = (size_t)blockIdx.x * 256 + threadIdx.x;   // < 3,670,016
    int e    = (int)(idx & 15);
    int lane = (int)((idx >> 4) & 31);
    int kt   = (int)((idx >> 9) % KT_C);
    int rest = (int)((idx >> 9) / KT_C);
    int ntile = rest & 127;
    int dir   = rest >> 7;
    int n = ntile * 16 + (lane & 15);
    int k = kt * 32 + (lane >> 4) * 8 + (e & 7) + (e >> 3) * 16;
    const float* wih = dir ? w_ih_b : w_ih_f;
    const float* whh = dir ? w_hh_b : w_hh_f;
    float v = (k < H_DIM) ? whh[(size_t)n * H_DIM + k]
                          : wih[(size_t)n * K_IN + (k - H_DIM)];
    wpack[idx] = (__bf16)v;
}

// ---------------------------------------------------------------------------
// Pack W_word into B-fragment order:
//   wwpack[((ntile*32 + kt)*32 + lane)*16 + e], ntile 0..63, kt 0..31
//   value = W_word[k][n], n = ntile*16 + lane%16, k = kt*32 + (lane/16)*8 + (e%8) + (e/8)*16
// ---------------------------------------------------------------------------
__global__ __launch_bounds__(256) void pack_wword_kernel(
    const float* __restrict__ W_word, __bf16* __restrict__ wwpack)
{
    size_t idx = (size_t)blockIdx.x * 256 + threadIdx.x;   // < 1,048,576
    int e    = (int)(idx & 15);
    int lane = (int)((idx >> 4) & 31);
    int kt   = (int)((idx >> 9) & 31);
    int ntile = (int)(idx >> 14);
    int n = ntile * 16 + (lane & 15);
    int k = kt * 32 + (lane >> 4) * 8 + (e & 7) + (e >> 3) * 16;
    wwpack[idx] = (__bf16)W_word[(size_t)k * TWOH + n];
}

// ---------------------------------------------------------------------------
// Embedding gather + concat c_embed -> bf16 x [T*B][384]
// ---------------------------------------------------------------------------
__global__ __launch_bounds__(384) void gather_x_kernel(
    const int* __restrict__ embed, const float* __restrict__ c_embed,
    const float* __restrict__ emb_table, __bf16* __restrict__ x_bf)
{
    int row = blockIdx.x;           // t*B + b
    int j = threadIdx.x;            // 0..383
    int tok = embed[row];
    float v = (j < E_DIM) ? emb_table[(size_t)tok * E_DIM + j]
                          : c_embed[(size_t)row * C_DIM + (j - E_DIM)];
    x_bf[(size_t)row * K_IN + j] = (__bf16)v;
}

// ---------------------------------------------------------------------------
// Init: h (bf16), c (f32), zero logits and the grid-barrier state.
// ---------------------------------------------------------------------------
__global__ __launch_bounds__(256) void init_state_kernel(
    const float* __restrict__ h0, const float* __restrict__ c0,
    __bf16* __restrict__ h_bf, float* __restrict__ cst,
    float* __restrict__ logits, unsigned* __restrict__ barrier_state)
{
    int idx = blockIdx.x * 256 + threadIdx.x;   // 0..65535
    if (idx < 2 * B_SZ * H_DIM) {
        h_bf[idx] = (__bf16)h0[idx];
        cst[idx] = c0[idx];
    }
    if (idx < T_LEN * B_SZ) logits[idx] = 0.0f;
    if (idx < 2) barrier_state[idx] = 0u;       // [0]=count, [1]=generation
}

// ---------------------------------------------------------------------------
// Device-scope grid barrier (counter + generation, s_sleep spin).
// ---------------------------------------------------------------------------
__device__ __forceinline__ void grid_sync(unsigned* bar, unsigned* gen, unsigned nb)
{
    __syncthreads();
    if (threadIdx.x == 0) {
        __threadfence();   // make this block's stores device-visible
        unsigned my = __hip_atomic_load(gen, __ATOMIC_RELAXED, __HIP_MEMORY_SCOPE_AGENT);
        unsigned arrived = __hip_atomic_fetch_add(bar, 1u, __ATOMIC_ACQ_REL,
                                                  __HIP_MEMORY_SCOPE_AGENT);
        if (arrived == nb - 1) {
            __hip_atomic_store(bar, 0u, __ATOMIC_RELAXED, __HIP_MEMORY_SCOPE_AGENT);
            __hip_atomic_fetch_add(gen, 1u, __ATOMIC_RELEASE, __HIP_MEMORY_SCOPE_AGENT);
        } else {
            while (__hip_atomic_load(gen, __ATOMIC_ACQUIRE,
                                     __HIP_MEMORY_SCOPE_AGENT) == my)
                __builtin_amdgcn_s_sleep(1);
        }
    }
    __syncthreads();
    __threadfence();       // acquire: invalidate stale lines before re-reading
}

// ---------------------------------------------------------------------------
// Persistent bidirectional LSTM scan. 128 blocks x 8 waves = 1024 waves;
// each wave owns one 16x16 tile of the (dir,64,2048) gate GEMM and keeps its
// entire K=896 weight stripe register-resident (28 x v16bf = 224 VGPRs,
// loaded once for all 512 timesteps). Per step:
//   phase A: gates = [h | x_t] @ Wcomb  (28 WMMAs, A = 2x b128 loads each)
//   grid barrier
//   phase B: LSTM gate math, update c/h, emit Hout (bf16)
//   grid barrier
// ---------------------------------------------------------------------------
__global__ __launch_bounds__(256, 1) void lstm_scan_kernel(
    __bf16* __restrict__ h_bf, const __bf16* __restrict__ x_bf,
    const __bf16* __restrict__ wpack, float* __restrict__ gates,
    const float* __restrict__ b_f, const float* __restrict__ b_b,
    float* __restrict__ cst, __bf16* __restrict__ hout_bf,
    unsigned* __restrict__ barrier_state)
{
    int tid = threadIdx.x;
    int lane = tid & 31;
    int wid = blockIdx.x * 8 + (tid >> 5);    // 0..1023
    int dir = wid >> 9;
    int rem = wid & 511;
    int mtile = rem >> 7;                     // 0..3
    int ntile = rem & 127;                    // 0..127

    int hi = (lane >> 4) & 1;
    int arow = mtile * 16 + (lane & 15);      // batch row
    int ncol = ntile * 16 + (lane & 15);      // gate column

    unsigned* bar = barrier_state;
    unsigned* gen = barrier_state + 1;
    unsigned nb = gridDim.x;

    // ---- one-time weight preload into registers ----
    v16bf bfrag[KT_C];
    const __bf16* wp = wpack + (size_t)((dir * 128 + ntile) * KT_C) * 32 * 16;
    #pragma unroll
    for (int kt = 0; kt < KT_C; ++kt) {
        const uint4* p = reinterpret_cast<const uint4*>(
            wp + ((size_t)kt * 32 + lane) * 16);
        FragU f; f.u[0] = p[0]; f.u[1] = p[1];
        bfrag[kt] = f.v;
    }

    const __bf16* hrow = h_bf + ((size_t)dir * B_SZ + arow) * H_DIM;
    float* gout = gates + (size_t)dir * B_SZ * G4;
    int gtid = blockIdx.x * 256 + tid;        // 0..32767

    for (int t = 0; t < T_LEN; ++t) {
        // ---------------- phase A: gate GEMM ----------------
        int tsrc = dir ? (T_LEN - 1 - t) : t;
        const __bf16* xrow = x_bf + ((size_t)tsrc * B_SZ + arow) * K_IN;

        v8f acc = {0.f, 0.f, 0.f, 0.f, 0.f, 0.f, 0.f, 0.f};
        #pragma unroll
        for (int kt = 0; kt < KT_C; ++kt) {
            int k0 = kt * 32;
            const __bf16* base = (kt < 16) ? (hrow + k0) : (xrow + (k0 - H_DIM));
            FragU fa;
            fa.u[0] = *reinterpret_cast<const uint4*>(base + 8 * hi);
            fa.u[1] = *reinterpret_cast<const uint4*>(base + 16 + 8 * hi);
            acc = __builtin_amdgcn_wmma_f32_16x16x32_bf16(
                      false, fa.v, false, bfrag[kt], (short)0, acc, false, false);
        }
        #pragma unroll
        for (int r = 0; r < 8; ++r)
            gout[(size_t)(mtile * 16 + r + 8 * hi) * G4 + ncol] = acc[r];

        grid_sync(bar, gen, nb);

        // ---------------- phase B: gate nonlinearity ----------------
        #pragma unroll
        for (int j = 0; j < 2; ++j) {
            int idx = gtid + 32768 * j;       // 0..65535
            int d2 = idx >> 15;
            int r2 = idx & 32767;
            int b = r2 >> 9;
            int h = r2 & 511;
            const float* bias = d2 ? b_b : b_f;
            const float* g = gates + ((size_t)d2 * B_SZ + b) * G4;

            float gi = g[h]             + bias[h];
            float gf = g[H_DIM + h]     + bias[H_DIM + h];
            float gg = g[2 * H_DIM + h] + bias[2 * H_DIM + h];
            float go = g[3 * H_DIM + h] + bias[3 * H_DIM + h];

            float c_new = sigf(gf) * cst[idx] + sigf(gi) * tanhf(gg);
            float h_new = sigf(go) * tanhf(c_new);

            cst[idx] = c_new;
            h_bf[idx] = (__bf16)h_new;
            int tout = d2 ? (T_LEN - 1 - t) : t;
            hout_bf[((size_t)tout * B_SZ + b) * TWOH + d2 * H_DIM + h] =
                (__bf16)h_new;
        }
        grid_sync(bar, gen, nb);
    }
}

// ---------------------------------------------------------------------------
// Fused squish+logits: logits[row] += sum_n tanh(Hout@W_word + b_word)*w_proj.
// One wave handles 4 N-tiles (A fragment reused 4x from registers); both A
// and packed-B fragments are 2x b128 loads. Epilogue: cross-lane reduce over
// the 16 columns of each tile, one f32 atomic per row half.
// ---------------------------------------------------------------------------
__global__ __launch_bounds__(256) void squish_logits_kernel(
    const __bf16* __restrict__ hout_bf, const __bf16* __restrict__ wwpack,
    const float* __restrict__ b_word, const float* __restrict__ w_proj,
    float* __restrict__ logits)
{
    int tid = threadIdx.x;
    int lane = tid & 31;
    int wid = blockIdx.x * 8 + (tid >> 5);    // 0..16383
    int mtile = wid >> 4;                     // 0..1023
    int ngroup = wid & 15;                    // 0..15 -> ntiles 4*ngroup..+3

    int hi = (lane >> 4) & 1;
    int arow = mtile * 16 + (lane & 15);
    const __bf16* Arow = hout_bf + (size_t)arow * TWOH;

    v8f z = {0.f, 0.f, 0.f, 0.f, 0.f, 0.f, 0.f, 0.f};
    v8f acc[4] = {z, z, z, z};

    for (int kt = 0; kt < 32; ++kt) {
        int k0 = kt * 32;
        FragU fa;
        fa.u[0] = *reinterpret_cast<const uint4*>(Arow + k0 + 8 * hi);
        fa.u[1] = *reinterpret_cast<const uint4*>(Arow + k0 + 16 + 8 * hi);
        #pragma unroll
        for (int i = 0; i < 4; ++i) {
            int nt = ngroup * 4 + i;
            const uint4* pb = reinterpret_cast<const uint4*>(
                wwpack + ((size_t)(nt * 32 + kt) * 32 + lane) * 16);
            FragU fb; fb.u[0] = pb[0]; fb.u[1] = pb[1];
            acc[i] = __builtin_amdgcn_wmma_f32_16x16x32_bf16(
                         false, fa.v, false, fb.v, (short)0, acc[i], false, false);
        }
    }

    int rowbase = mtile * 16;
    #pragma unroll
    for (int i = 0; i < 4; ++i) {
        int gcol = (ngroup * 4 + i) * 16 + (lane & 15);
        float bw = b_word[gcol];
        float wpj = w_proj[gcol];
        #pragma unroll
        for (int r = 0; r < 8; ++r) {
            float v = tanhf(acc[i][r] + bw) * wpj;
            #pragma unroll
            for (int m = 1; m < 16; m <<= 1) v += __shfl_xor(v, m, 32);
            if ((lane & 15) == 0)
                atomicAdd(&logits[rowbase + r + 8 * hi], v);
        }
    }
}

// ---------------------------------------------------------------------------
__global__ __launch_bounds__(256) void softmax_kernel(
    const float* __restrict__ logits, float* __restrict__ attn)
{
    __shared__ float red[256];
    int b = blockIdx.x, tid = threadIdx.x;
    float m = -1e30f;
    for (int t = tid; t < T_LEN; t += 256) m = fmaxf(m, logits[t * B_SZ + b]);
    red[tid] = m; __syncthreads();
    for (int s = 128; s > 0; s >>= 1) {
        if (tid < s) red[tid] = fmaxf(red[tid], red[tid + s]);
        __syncthreads();
    }
    float mx = red[0]; __syncthreads();
    float sum = 0.f;
    for (int t = tid; t < T_LEN; t += 256) sum += expf(logits[t * B_SZ + b] - mx);
    red[tid] = sum; __syncthreads();
    for (int s = 128; s > 0; s >>= 1) {
        if (tid < s) red[tid] += red[tid + s];
        __syncthreads();
    }
    float inv = 1.0f / red[0];
    for (int t = tid; t < T_LEN; t += 256)
        attn[t * B_SZ + b] = expf(logits[t * B_SZ + b] - mx) * inv;
}

// ---------------------------------------------------------------------------
__global__ __launch_bounds__(256) void ctx_kernel(
    const float* __restrict__ attn, const __bf16* __restrict__ hout_bf,
    float* __restrict__ ctx)
{
    __shared__ float sa[T_LEN];
    int b = blockIdx.x, tid = threadIdx.x;
    for (int t = tid; t < T_LEN; t += 256) sa[t] = attn[t * B_SZ + b];
    __syncthreads();
    #pragma unroll
    for (int j = 0; j < 4; ++j) {
        int d = tid + 256 * j;
        float acc = 0.f;
        for (int t = 0; t < T_LEN; ++t)
            acc += sa[t] * (float)hout_bf[((size_t)t * B_SZ + b) * TWOH + d];
        ctx[(size_t)b * TWOH + d] = acc;
    }
}

// ---------------------------------------------------------------------------
__global__ __launch_bounds__(320) void final_kernel(
    const float* __restrict__ ctx, const float* __restrict__ lin_W,
    const float* __restrict__ lin_b, float* __restrict__ out)
{
    int tid = threadIdx.x;
    if (tid >= B_SZ * NC) return;
    int b = tid / NC, n = tid % NC;
    float s = 0.f;
    for (int d = 0; d < TWOH; ++d)
        s += ctx[(size_t)b * TWOH + d] * lin_W[(size_t)n * TWOH + d];
    out[b * NC + n] = s + lin_b[n];
}

// ---------------------------------------------------------------------------
extern "C" void kernel_launch(void* const* d_in, const int* in_sizes, int n_in,
                              void* d_out, int out_size, void* d_ws, size_t ws_size,
                              hipStream_t stream) {
    const int*   embed     = (const int*)d_in[0];
    const float* c_embed   = (const float*)d_in[1];
    const float* h0        = (const float*)d_in[2];
    const float* c0        = (const float*)d_in[3];
    const float* emb_table = (const float*)d_in[4];
    const float* w_ih_f    = (const float*)d_in[5];
    const float* w_hh_f    = (const float*)d_in[6];
    const float* b_f       = (const float*)d_in[7];
    const float* w_ih_b    = (const float*)d_in[8];
    const float* w_hh_b    = (const float*)d_in[9];
    const float* b_b       = (const float*)d_in[10];
    const float* W_word    = (const float*)d_in[11];
    const float* b_word    = (const float*)d_in[12];
    const float* w_proj    = (const float*)d_in[13];
    const float* lin_W     = (const float*)d_in[14];
    const float* lin_b     = (const float*)d_in[15];
    float* out = (float*)d_out;

    char* ws = (char*)d_ws;
    size_t off = 0;
    __bf16* x_bf    = (__bf16*)(ws + off); off += (size_t)T_LEN * B_SZ * K_IN * 2;  // 25.2 MB
    __bf16* wpack   = (__bf16*)(ws + off); off += (size_t)2 * K_C * G4 * 2;         // 7.3 MB
    __bf16* wwpack  = (__bf16*)(ws + off); off += (size_t)TWOH * TWOH * 2;          // 2.1 MB
    __bf16* hout_bf = (__bf16*)(ws + off); off += (size_t)T_LEN * B_SZ * TWOH * 2;  // 67.1 MB
    __bf16* h_bf    = (__bf16*)(ws + off); off += (size_t)2 * B_SZ * H_DIM * 2;
    float*  cst     = (float*)(ws + off);  off += (size_t)2 * B_SZ * H_DIM * 4;
    float*  gates   = (float*)(ws + off);  off += (size_t)2 * B_SZ * G4 * 4;
    float*  logits  = (float*)(ws + off);  off += (size_t)T_LEN * B_SZ * 4;
    float*  attn    = (float*)(ws + off);  off += (size_t)T_LEN * B_SZ * 4;
    float*  ctx     = (float*)(ws + off);  off += (size_t)B_SZ * TWOH * 4;
    unsigned* barst = (unsigned*)(ws + off); off += 64;   // barrier count + gen

    pack_wcomb_kernel<<<(2 * K_C * G4) / 256, 256, 0, stream>>>(
        w_ih_f, w_hh_f, w_ih_b, w_hh_b, wpack);
    pack_wword_kernel<<<(TWOH * TWOH) / 256, 256, 0, stream>>>(W_word, wwpack);
    gather_x_kernel<<<T_LEN * B_SZ, K_IN, 0, stream>>>(embed, c_embed, emb_table, x_bf);
    init_state_kernel<<<(2 * B_SZ * H_DIM) / 256, 256, 0, stream>>>(
        h0, c0, h_bf, cst, logits, barst);

    lstm_scan_kernel<<<NBLK, 256, 0, stream>>>(
        h_bf, x_bf, wpack, gates, b_f, b_b, cst, hout_bf, barst);

    squish_logits_kernel<<<(T_LEN * B_SZ / 32) * 16 / 8, 256, 0, stream>>>(
        hout_bf, wwpack, b_word, w_proj, logits);
    softmax_kernel<<<B_SZ, 256, 0, stream>>>(logits, attn);
    ctx_kernel<<<B_SZ, 256, 0, stream>>>(attn, hout_bf, ctx);
    final_kernel<<<1, 320, 0, stream>>>(ctx, lin_W, lin_b, out);
}
